// DummyMPNN_46840913330618
// MI455X (gfx1250) — matrix operand order repaired and checked by reading, not compiled, not run on hardware
//
#include <hip/hip_runtime.h>
#include <math.h>

#define NNODE 10000
#define NEDGE 30000
#define NB    256
#define DIM   64
#define NF    14
#define EF    4
#define ETILES 1875      // NEDGE / 16
#define NCT    256       // 4096 / 16 column tiles
#define NEG_SLOPE 0.01f

typedef __attribute__((ext_vector_type(16))) __bf16 v16bf;
typedef __attribute__((ext_vector_type(8)))  float  v8f;

__device__ __forceinline__ float leaky_(float v) { return v > 0.f ? v : NEG_SLOPE * v; }
__device__ __forceinline__ float sigm_(float v)  { return 1.f / (1.f + __expf(-v)); }
__device__ __forceinline__ unsigned short f2bf_(float f) {
    union { float f; unsigned u; } x; x.f = f;
    unsigned u = x.u + 0x7FFFu + ((x.u >> 16) & 1u);   // round-to-nearest-even
    return (unsigned short)(u >> 16);
}

// ---------------------------------------------------------------- utility
__global__ void mpnn_zero_kernel(float* p, int n) {
    int t = blockIdx.x * blockDim.x + threadIdx.x;
    if (t < n) p[t] = 0.f;
}

// ------------------------------------------------- pack net_w2^T into WMMA-B layout (bf16)
// B tile for GEMM: B[k, c] = net_w2[c*64 + k]; layout per (ct, kt):
// lane<16: column n=lane, K = kt*32 + {0..7, 16..23}; lane>=16: n=lane-16, K = kt*32 + {8..15, 24..31}
// element j holds K = base + (j>>3)*16 + (j&7).
__global__ void mpnn_pack_w2_kernel(const float* __restrict__ w2,
                                    unsigned short* __restrict__ w2p) {
    int tid = blockIdx.x * blockDim.x + threadIdx.x;     // 16384 threads
    int ct = tid >> 6, rem = tid & 63, kt = rem >> 5, lane = rem & 31;
    int n = lane & 15, c = ct * 16 + n;
    int kbase = kt * 32 + ((lane >> 4) << 3);
    unsigned short tmp[16];
#pragma unroll
    for (int j = 0; j < 16; ++j) {
        int v = j >> 1, p = j & 1;
        int k = kbase + ((j >> 3) << 4) + (j & 7);
        tmp[j] = f2bf_(w2[c * 64 + k]);
    }
    unsigned short* dst = w2p + ((size_t)(ct * 2 + kt) * 32 + lane) * 16;
    ((uint4*)dst)[0] = ((uint4*)tmp)[0];
    ((uint4*)dst)[1] = ((uint4*)tmp)[1];
}

// ------------------------------------------------- edge_h = leaky(edge_attr @ w1^T + b1), packed as WMMA-A
// A tile (16 edges x 32 K): lane row m = lane&15;
// lane<16:  K = kt*32 + {0..7 (v0..3), 16..23 (v4..7)}
// lane>=16: K = kt*32 + {8..15, 24..31}
__global__ void mpnn_pack_A_kernel(const float* __restrict__ ea,
                                   const float* __restrict__ w1,
                                   const float* __restrict__ b1,
                                   unsigned short* __restrict__ eAp) {
    int tid = blockIdx.x * blockDim.x + threadIdx.x;     // 120000 threads
    if (tid >= ETILES * 2 * 32) return;
    int etile = tid >> 6, rem = tid & 63, kt = rem >> 5, lane = rem & 31;
    int m = lane & 15;
    int e = etile * 16 + m;
    float a0 = ea[e * EF + 0], a1 = ea[e * EF + 1], a2 = ea[e * EF + 2], a3 = ea[e * EF + 3];
    unsigned short tmp[16];
#pragma unroll
    for (int j = 0; j < 16; ++j) {
        int v = j >> 1, p = j & 1;
        int k = kt * 32 + ((v >> 2) << 4) + ((lane >> 4) << 3) + ((v & 3) << 1) + p;
        float h = a0 * w1[k * EF + 0] + a1 * w1[k * EF + 1] +
                  a2 * w1[k * EF + 2] + a3 * w1[k * EF + 3] + b1[k];
        tmp[j] = f2bf_(leaky_(h));
    }
    unsigned short* dst = eAp + ((size_t)(etile * 2 + kt) * 32 + lane) * 16;
    ((uint4*)dst)[0] = ((uint4*)tmp)[0];
    ((uint4*)dst)[1] = ((uint4*)tmp)[1];
}

// ------------------------------------------------- out = leaky(x @ lin0_w^T + b)
__global__ void mpnn_lin0_kernel(const float* __restrict__ x,
                                 const float* __restrict__ w,
                                 const float* __restrict__ b,
                                 float* __restrict__ out) {
    int tid = blockIdx.x * blockDim.x + threadIdx.x;     // N*64
    if (tid >= NNODE * DIM) return;
    int n = tid >> 6, d = tid & 63;
    float acc = b[d];
#pragma unroll
    for (int f = 0; f < NF; ++f) acc += x[n * NF + f] * w[d * NF + f];
    out[tid] = leaky_(acc);
}

// ------------------------------------------------- degree (scatter count on dst)
__global__ void mpnn_deg_kernel(const int* __restrict__ dst, float* __restrict__ deg) {
    int e = blockIdx.x * blockDim.x + threadIdx.x;
    if (e < NEDGE) atomicAdd(&deg[dst[e]], 1.0f);
}

// ------------------------------------------------- fused WMMA message kernel
// For each 16-edge tile: recompute W tile = edge_h_tile @ net_w2^T (+b2) via
// v_wmma_f32_16x16x32_bf16 and immediately contract with out[src] -> msg,
// scatter-add into agg via atomics. 8 waves / block, B operand staged in LDS.
__launch_bounds__(256)
__global__ void mpnn_msg_kernel(const unsigned short* __restrict__ eAp,
                                const unsigned short* __restrict__ w2p,
                                const float* __restrict__ b2,
                                const float* __restrict__ out,
                                const int* __restrict__ src,
                                const int* __restrict__ dstI,
                                float* __restrict__ agg) {
    __shared__ unsigned short bop[8 * 1024];   // 8 ct-blocks * 1024 bf16 = 16KB
    __shared__ float outl[8 * 16 * 64];        // 8 waves * 16 edges * 64 = 32KB

    int tid  = threadIdx.x;
    int wave = tid >> 5, lane = tid & 31;
    int etile = blockIdx.x * 8 + wave;
    bool valid = etile < ETILES;

    // stage the 16 source-node feature vectors for this wave's edge tile
    if (valid) {
        for (int r = 0; r < 32; ++r) {
            int idx = lane + r * 32;           // 0..1023
            int m = idx >> 6, f = idx & 63;
            int e = etile * 16 + m;
            outl[wave * 1024 + idx] = out[(size_t)src[e] * DIM + f];
        }
    }

    v16bf a0{}, a1{};
    if (valid) {
        a0 = *(const v16bf*)(eAp + ((size_t)(etile * 2 + 0) * 32 + lane) * 16);
        a1 = *(const v16bf*)(eAp + ((size_t)(etile * 2 + 1) * 32 + lane) * 16);
    }

    int n  = lane & 15;
    int mb = (lane >> 4) << 3;                  // 0 or 8
    float macc[32];
#pragma unroll
    for (int i = 0; i < 32; ++i) macc[i] = 0.f;

    const uint4* w2p4 = (const uint4*)w2p;
    for (int chunk = 0; chunk < 32; ++chunk) {  // 8 ct per chunk
        __syncthreads();
        for (int t = tid; t < 1024; t += 256)
            ((uint4*)bop)[t] = w2p4[chunk * 1024 + t];
        __syncthreads();
        if (valid) {
#pragma unroll
            for (int ii = 0; ii < 2; ++ii) {
                int i = chunk * 2 + ii;         // K index of `out`
                float s[8];
#pragma unroll
                for (int v = 0; v < 8; ++v) s[v] = outl[wave * 1024 + (mb + v) * 64 + i];
#pragma unroll
                for (int q = 0; q < 4; ++q) {
                    int ct  = i * 4 + q;        // column tile 0..255
                    int lct = ii * 4 + q;       // within staged chunk
                    v16bf b0v = *(const v16bf*)(bop + lct * 1024 + lane * 16);
                    v16bf b1v = *(const v16bf*)(bop + lct * 1024 + 512 + lane * 16);
                    float bias = b2[ct * 16 + n];
                    v8f c = {};
                    c = __builtin_amdgcn_wmma_f32_16x16x32_bf16(false, a0, false, b0v,
                                                                (short)0, c, false, false);
                    c = __builtin_amdgcn_wmma_f32_16x16x32_bf16(false, a1, false, b1v,
                                                                (short)0, c, false, false);
#pragma unroll
                    for (int v = 0; v < 8; ++v)
                        macc[v * 4 + q] += (c[v] + bias) * s[v];
                }
            }
        }
    }

    if (valid) {
#pragma unroll
        for (int v = 0; v < 8; ++v) {
            int e = etile * 16 + (mb + v);
            size_t base = (size_t)dstI[e] * DIM;
#pragma unroll
            for (int q = 0; q < 4; ++q) {
                int o = q * 16 + n;
                atomicAdd(&agg[base + o], macc[v * 4 + q]);
            }
        }
    }
}

// ------------------------------------------------- node update: NNConv root + GRU step
__launch_bounds__(256)
__global__ void mpnn_node_kernel(float* __restrict__ out,
                                 const float* __restrict__ agg,
                                 const float* __restrict__ deg,
                                 const float* __restrict__ conv_root,
                                 const float* __restrict__ conv_bias,
                                 const float* __restrict__ gw_ih,
                                 const float* __restrict__ gw_hh,
                                 const float* __restrict__ gb_ih,
                                 const float* __restrict__ gb_hh) {
    __shared__ float sh[4 * 64];   // old out / h
    __shared__ float shm[4 * 64];  // m
    int tid = threadIdx.x;
    int g = tid >> 6, d = tid & 63;
    int node = blockIdx.x * 4 + g;               // 2500 blocks * 4 == NNODE exactly

    float ov = out[(size_t)node * DIM + d];
    sh[g * 64 + d] = ov;
    __syncthreads();

    float a = agg[(size_t)node * DIM + d] / fmaxf(deg[node], 1.0f);
    float cr = conv_bias[d];
#pragma unroll 8
    for (int i = 0; i < DIM; ++i) cr += sh[g * 64 + i] * conv_root[d * DIM + i];
    float mval = leaky_(a + cr);
    shm[g * 64 + d] = mval;
    __syncthreads();

    float gi_r = gb_ih[d], gi_z = gb_ih[64 + d], gi_n = gb_ih[128 + d];
    float gh_r = gb_hh[d], gh_z = gb_hh[64 + d], gh_n = gb_hh[128 + d];
#pragma unroll 8
    for (int i = 0; i < DIM; ++i) {
        float mi = shm[g * 64 + i];
        float hi = sh[g * 64 + i];
        gi_r += mi * gw_ih[(0 * 64 + d) * DIM + i];
        gi_z += mi * gw_ih[(64 + d) * DIM + i];
        gi_n += mi * gw_ih[(128 + d) * DIM + i];
        gh_r += hi * gw_hh[(0 * 64 + d) * DIM + i];
        gh_z += hi * gw_hh[(64 + d) * DIM + i];
        gh_n += hi * gw_hh[(128 + d) * DIM + i];
    }
    float r = sigm_(gi_r + gh_r);
    float z = sigm_(gi_z + gh_z);
    float nn = tanhf(gi_n + r * gh_n);
    float h = (1.f - z) * nn + z * ov;
    out[(size_t)node * DIM + d] = h;
}

// ------------------------------------------------- Set2Set query q (bias-only LSTM step)
__global__ void mpnn_q_kernel(const float* __restrict__ b_ih,
                              const float* __restrict__ b_hh,
                              float* __restrict__ qvec) {
    int d = threadIdx.x;                       // 64 threads
    float gi = b_ih[d] + b_hh[d];
    float gf = b_ih[64 + d] + b_hh[64 + d];
    float gg = b_ih[128 + d] + b_hh[128 + d];
    float go = b_ih[192 + d] + b_hh[192 + d];
    float c = sigm_(gi) * tanhf(gg);           // f*0 + i*g
    (void)gf;
    qvec[d] = sigm_(go) * tanhf(c);
}

// ------------------------------------------------- per-graph softmax attention + output linear
__launch_bounds__(256)
__global__ void mpnn_final_kernel(const float* __restrict__ out,
                                  const int* __restrict__ batch,
                                  const float* __restrict__ qvec,
                                  float* __restrict__ e_arr,
                                  const float* __restrict__ low,
                                  const float* __restrict__ lob,
                                  float* __restrict__ y) {
    __shared__ float qsh[64];
    __shared__ float red[256];
    __shared__ float rsh[64];
    __shared__ float wsumsh;
    __shared__ float emaxsh;

    int b = blockIdx.x, tid = threadIdx.x;
    if (tid < 64) { qsh[tid] = qvec[tid]; rsh[tid] = 0.f; }
    if (tid == 0) wsumsh = 0.f;
    __syncthreads();

    // lower_bound over sorted batch[]
    int start, end;
    {
        int lo = 0, hi = NNODE;
        while (lo < hi) { int mid = (lo + hi) >> 1; if (batch[mid] < b) lo = mid + 1; else hi = mid; }
        start = lo;
        lo = start; hi = NNODE;
        while (lo < hi) { int mid = (lo + hi) >> 1; if (batch[mid] < b + 1) lo = mid + 1; else hi = mid; }
        end = lo;
    }

    float lm = -3.4e38f;
    for (int nn = start + tid; nn < end; nn += 256) {
        float e = 0.f;
#pragma unroll 8
        for (int d = 0; d < DIM; ++d) e += out[(size_t)nn * DIM + d] * qsh[d];
        e_arr[nn] = e;
        lm = fmaxf(lm, e);
    }
    red[tid] = lm;
    __syncthreads();
    for (int s = 128; s > 0; s >>= 1) {
        if (tid < s) red[tid] = fmaxf(red[tid], red[tid + s]);
        __syncthreads();
    }
    if (tid == 0) emaxsh = red[0];
    __syncthreads();
    float emax = emaxsh;

    for (int nn = start + tid; nn < end; nn += 256) {
        float w = __expf(e_arr[nn] - emax);
        atomicAdd(&wsumsh, w);
#pragma unroll 8
        for (int d = 0; d < DIM; ++d)
            atomicAdd(&rsh[d], w * out[(size_t)nn * DIM + d]);
    }
    __syncthreads();

    if (tid < 2) {
        float inv = wsumsh > 0.f ? 1.f / wsumsh : 0.f;
        float acc = lob[tid];
#pragma unroll 8
        for (int d = 0; d < DIM; ++d) {
            acc += qsh[d] * low[tid * 128 + d];
            acc += (rsh[d] * inv) * low[tid * 128 + 64 + d];
        }
        y[b * 2 + tid] = acc;
    }
}

// ================================================================ host side
extern "C" void kernel_launch(void* const* d_in, const int* in_sizes, int n_in,
                              void* d_out, int out_size, void* d_ws, size_t ws_size,
                              hipStream_t stream) {
    const float* x         = (const float*)d_in[0];
    const int*   edge_idx  = (const int*)d_in[1];
    const float* edge_attr = (const float*)d_in[2];
    const int*   batch     = (const int*)d_in[3];
    const float* lin0_w    = (const float*)d_in[4];
    const float* lin0_b    = (const float*)d_in[5];
    const float* net_w1    = (const float*)d_in[6];
    const float* net_b1    = (const float*)d_in[7];
    const float* net_w2    = (const float*)d_in[8];
    const float* net_b2    = (const float*)d_in[9];
    const float* conv_root = (const float*)d_in[10];
    const float* conv_bias = (const float*)d_in[11];
    const float* gru_w_ih  = (const float*)d_in[12];
    const float* gru_w_hh  = (const float*)d_in[13];
    const float* gru_b_ih  = (const float*)d_in[14];
    const float* gru_b_hh  = (const float*)d_in[15];
    const float* lstm_b_ih = (const float*)d_in[18];
    const float* lstm_b_hh = (const float*)d_in[19];
    const float* lin_out_w = (const float*)d_in[20];
    const float* lin_out_b = (const float*)d_in[21];
    float* y = (float*)d_out;

    const int* src = edge_idx;            // edge_index[0]
    const int* dst = edge_idx + NEDGE;    // edge_index[1]

    // workspace carve (256B aligned)
    char* base = (char*)d_ws;
    size_t off = 0;
    auto carve = [&](size_t bytes) { char* p = base + off; off = (off + bytes + 255) & ~(size_t)255; return p; };
    unsigned short* w2p  = (unsigned short*)carve((size_t)NCT * 2 * 32 * 16 * 2);      // 512 KB
    unsigned short* eAp  = (unsigned short*)carve((size_t)ETILES * 2 * 32 * 16 * 2);   // 3.84 MB
    float* outbuf = (float*)carve((size_t)NNODE * DIM * 4);
    float* aggbuf = (float*)carve((size_t)NNODE * DIM * 4);
    float* deg    = (float*)carve((size_t)NNODE * 4);
    float* e_arr  = (float*)carve((size_t)NNODE * 4);
    float* qvec   = (float*)carve(256);
    (void)ws_size; (void)in_sizes; (void)n_in; (void)out_size;

    // one-time precompute
    mpnn_pack_w2_kernel<<<64, 256, 0, stream>>>(net_w2, w2p);
    mpnn_pack_A_kernel<<<(ETILES * 64 + 255) / 256, 256, 0, stream>>>(edge_attr, net_w1, net_b1, eAp);
    mpnn_lin0_kernel<<<(NNODE * DIM + 255) / 256, 256, 0, stream>>>(x, lin0_w, lin0_b, outbuf);
    mpnn_zero_kernel<<<(NNODE + 255) / 256, 256, 0, stream>>>(deg, NNODE);
    mpnn_deg_kernel<<<(NEDGE + 255) / 256, 256, 0, stream>>>(dst, deg);

    // 6 message-passing iterations
    for (int it = 0; it < 6; ++it) {
        mpnn_zero_kernel<<<(NNODE * DIM + 255) / 256, 256, 0, stream>>>(aggbuf, NNODE * DIM);
        mpnn_msg_kernel<<<(ETILES + 7) / 8, 256, 0, stream>>>(eAp, w2p, net_b2, outbuf, src, dst, aggbuf);
        mpnn_node_kernel<<<NNODE / 4, 256, 0, stream>>>(outbuf, aggbuf, deg, conv_root, conv_bias,
                                                        gru_w_ih, gru_w_hh, gru_b_ih, gru_b_hh);
    }

    // Set2Set (1 step) + output
    mpnn_q_kernel<<<1, 64, 0, stream>>>(lstm_b_ih, lstm_b_hh, qvec);
    mpnn_final_kernel<<<NB, 256, 0, stream>>>(outbuf, batch, qvec, e_arr, lin_out_w, lin_out_b, y);
}
